// TurboQuantProd_44255343018361
// MI455X (gfx1250) — matrix operand order repaired and checked by reading, not compiled, not run on hardware
//
#include <hip/hip_runtime.h>
#include <math.h>

#define DIMV   4096
#define NROWSV 8192
#define KOUT   327                      // int(4096 * 0.08)
#define QJL_SCALE (1.2533141373155003f / 4096.0f)   // sqrt(pi/2)/DIM

typedef _Float16 v16h __attribute__((ext_vector_type(16)));
typedef _Float16 v8h  __attribute__((ext_vector_type(8)));
typedef float    v8f  __attribute__((ext_vector_type(8)));

__device__ __forceinline__ v16h frag16(const _Float16* p0, const _Float16* p1) {
  v8h lo = *(const v8h*)p0;
  v8h hi = *(const v8h*)p1;
  return __builtin_shufflevector(lo, hi, 0,1,2,3,4,5,6,7,8,9,10,11,12,13,14,15);
}

__device__ __forceinline__ float waveReduceSum(float s) {
  for (int off = 16; off > 0; off >>= 1) s += __shfl_down(s, off, 32);
  return s;
}

// ---------------------------------------------------------------------------
// WMMA GEMM: C[M,N] = A[M,K] * Bt[N,K]^T   (A, Bt fp16 row-major, K contiguous)
// Block: 256 threads = 8 waves. Wave tile 32(M) x 64(N)  -> block tile 256 x 64.
// Per K=32 step: 12 x global_load_b128 feed 8 x v_wmma_f32_16x16x32_f16.
// ---------------------------------------------------------------------------
__global__ void __launch_bounds__(256)
gemm_wmma_f16(const _Float16* __restrict__ A, const _Float16* __restrict__ Bt,
              float* __restrict__ C, int M, int Nn, int K) {
  const int lane = threadIdx.x & 31;
  const int wave = threadIdx.x >> 5;
  const int mBase = blockIdx.y * 256 + wave * 32;   // 32 rows per wave
  const int nBase = blockIdx.x * 64;
  if (mBase >= M) return;                           // wave-uniform

  // A fragment addressing (contiguous k order, same as B; not the ISA table order):
  //   lanes 0-15  : row = lane,     elements 0..15 -> K 0..15
  //   lanes 16-31 : row = lane-16,  elements 0..15 -> K 16..31
  const int kHalf = (lane >> 4) ? 16 : 0;
  // B fragment addressing (K x 16): lanes 0-15 cover K 0..15, lanes 16-31 K 16..31
  const int kOffB = (lane >> 4) ? 16 : 0;

  const _Float16* aRow0 = A + (size_t)(mBase +      (lane & 15)) * K + kHalf;
  const _Float16* aRow1 = A + (size_t)(mBase + 16 + (lane & 15)) * K + kHalf;
  const _Float16* b0 = Bt + (size_t)(nBase +  0 + (lane & 15)) * K + kOffB;
  const _Float16* b1 = Bt + (size_t)(nBase + 16 + (lane & 15)) * K + kOffB;
  const _Float16* b2 = Bt + (size_t)(nBase + 32 + (lane & 15)) * K + kOffB;
  const _Float16* b3 = Bt + (size_t)(nBase + 48 + (lane & 15)) * K + kOffB;

  v8f a00 = {}, a01 = {}, a02 = {}, a03 = {};   // rows mBase..mBase+15
  v8f a10 = {}, a11 = {}, a12 = {}, a13 = {};   // rows mBase+16..mBase+31

  for (int k = 0; k < K; k += 32) {
    v16h af0 = frag16(aRow0 + k, aRow0 + k + 8);
    v16h af1 = frag16(aRow1 + k, aRow1 + k + 8);
    v16h bf0 = frag16(b0 + k,    b0 + k + 8);
    v16h bf1 = frag16(b1 + k,    b1 + k + 8);
    v16h bf2 = frag16(b2 + k,    b2 + k + 8);
    v16h bf3 = frag16(b3 + k,    b3 + k + 8);
    a00 = __builtin_amdgcn_wmma_f32_16x16x32_f16(false, af0, false, bf0, (short)0, a00, false, false);
    a01 = __builtin_amdgcn_wmma_f32_16x16x32_f16(false, af0, false, bf1, (short)0, a01, false, false);
    a02 = __builtin_amdgcn_wmma_f32_16x16x32_f16(false, af0, false, bf2, (short)0, a02, false, false);
    a03 = __builtin_amdgcn_wmma_f32_16x16x32_f16(false, af0, false, bf3, (short)0, a03, false, false);
    a10 = __builtin_amdgcn_wmma_f32_16x16x32_f16(false, af1, false, bf0, (short)0, a10, false, false);
    a11 = __builtin_amdgcn_wmma_f32_16x16x32_f16(false, af1, false, bf1, (short)0, a11, false, false);
    a12 = __builtin_amdgcn_wmma_f32_16x16x32_f16(false, af1, false, bf2, (short)0, a12, false, false);
    a13 = __builtin_amdgcn_wmma_f32_16x16x32_f16(false, af1, false, bf3, (short)0, a13, false, false);
  }

  // C/D 16x16 f32 layout: lane<16 -> VGPR v = C[v][lane]; lane>=16 -> C[v+8][lane-16]
  const int col  = lane & 15;
  const int rOff = (lane >> 4) * 8;
  for (int v = 0; v < 8; ++v) {
    size_t r0 = (size_t)(mBase +      rOff + v) * Nn + nBase;
    size_t r1 = (size_t)(mBase + 16 + rOff + v) * Nn + nBase;
    C[r0 +  0 + col] = a00[v];
    C[r0 + 16 + col] = a01[v];
    C[r0 + 32 + col] = a02[v];
    C[r0 + 48 + col] = a03[v];
    C[r1 +  0 + col] = a10[v];
    C[r1 + 16 + col] = a11[v];
    C[r1 + 32 + col] = a12[v];
    C[r1 + 48 + col] = a13[v];
  }
}

// ---------------------------------------------------------------------------
// Elementwise / reduction kernels
// ---------------------------------------------------------------------------
__global__ void __launch_bounds__(256)
rownorm_kernel(const float* __restrict__ x, float* __restrict__ norms,
               _Float16* __restrict__ xu) {
  __shared__ float red[8];
  const int row = blockIdx.x;
  const float* xr = x + (size_t)row * DIMV;
  float s = 0.f;
  for (int j = threadIdx.x; j < DIMV; j += 256) { float v = xr[j]; s += v * v; }
  s = waveReduceSum(s);
  if ((threadIdx.x & 31) == 0) red[threadIdx.x >> 5] = s;
  __syncthreads();
  if (threadIdx.x == 0) {
    float t = 0.f;
    for (int w = 0; w < 8; ++w) t += red[w];
    float n = sqrtf(t);
    norms[row] = n;
    red[0] = 1.0f / (n + 1e-10f);
  }
  __syncthreads();
  const float inv = red[0];
  _Float16* xur = xu + (size_t)row * DIMV;
  for (int j = threadIdx.x; j < DIMV; j += 256) xur[j] = (_Float16)(xr[j] * inv);
}

__global__ void colvar_kernel(const float* __restrict__ x, float* __restrict__ var) {
  const int c = blockIdx.x * blockDim.x + threadIdx.x;
  if (c >= DIMV) return;
  float s = 0.f, ss = 0.f;
  for (int r = 0; r < NROWSV; ++r) {
    float v = x[(size_t)r * DIMV + c];
    s += v; ss += v * v;
  }
  float mean = s / (float)NROWSV;
  var[c] = (ss - mean * s) / (float)(NROWSV - 1);
}

__global__ void outlier_mask_kernel(const float* __restrict__ var, int* __restrict__ mask) {
  const int c = blockIdx.x * blockDim.x + threadIdx.x;
  if (c >= DIMV) return;
  const float vc = var[c];
  int rank = 0;
  for (int d = 0; d < DIMV; ++d) {
    float vd = var[d];
    rank += (vd > vc) || (vd == vc && d < c);   // stable argsort(-var) rank
  }
  mask[c] = (rank < KOUT) ? 1 : 0;
}

__global__ void quantize_kernel(const float* __restrict__ y,
                                const float* __restrict__ cent,
                                _Float16* __restrict__ yhat) {
  const float c0 = cent[0], c1 = cent[1], c2 = cent[2], c3 = cent[3];
  const float b0 = 0.5f * (c0 + c1), b1 = 0.5f * (c1 + c2), b2 = 0.5f * (c2 + c3);
  const size_t total = (size_t)NROWSV * DIMV;
  const size_t stride = (size_t)gridDim.x * blockDim.x;
  for (size_t i = (size_t)blockIdx.x * blockDim.x + threadIdx.x; i < total; i += stride) {
    float v = y[i];
    int idx = (v > b0) + (v > b1) + (v > b2);   // searchsorted(db, v)
    float q = (idx == 0) ? c0 : (idx == 1) ? c1 : (idx == 2) ? c2 : c3;
    yhat[i] = (_Float16)q;
  }
}

__global__ void __launch_bounds__(256)
residual_kernel(const float* __restrict__ x, const float* __restrict__ xhat,
                const float* __restrict__ norms, const int* __restrict__ mask,
                _Float16* __restrict__ res, float* __restrict__ resnorm) {
  __shared__ float red[8];
  const int row = blockIdx.x;
  const float nrm = norms[row];
  const float inv = 1.0f / (nrm + 1e-10f);
  const float* xr = x    + (size_t)row * DIMV;
  const float* hr = xhat + (size_t)row * DIMV;
  _Float16*    rr = res  + (size_t)row * DIMV;
  float s = 0.f;
  for (int j = threadIdx.x; j < DIMV; j += 256) {
    float xv = xr[j];
    float xh = mask[j] ? (xv * inv) : hr[j];     // outlier pass-through (unit domain)
    float rv = xv - xh * nrm;
    rr[j] = (_Float16)rv;
    s += rv * rv;
  }
  s = waveReduceSum(s);
  if ((threadIdx.x & 31) == 0) red[threadIdx.x >> 5] = s;
  __syncthreads();
  if (threadIdx.x == 0) {
    float t = 0.f;
    for (int w = 0; w < 8; ++w) t += red[w];
    resnorm[row] = sqrtf(t);
  }
}

__global__ void sign_kernel(const float* __restrict__ proj, _Float16* __restrict__ sg) {
  const size_t total = (size_t)NROWSV * DIMV;
  const size_t stride = (size_t)gridDim.x * blockDim.x;
  for (size_t i = (size_t)blockIdx.x * blockDim.x + threadIdx.x; i < total; i += stride)
    sg[i] = (proj[i] > 0.f) ? (_Float16)1.0f : (_Float16)(-1.0f);
}

__global__ void combine_kernel(const float* __restrict__ xhat, const float* __restrict__ g4,
                               const float* __restrict__ norms, const float* __restrict__ resnorm,
                               float* __restrict__ out) {
  const size_t total = (size_t)NROWSV * DIMV;
  const size_t stride = (size_t)gridDim.x * blockDim.x;
  for (size_t i = (size_t)blockIdx.x * blockDim.x + threadIdx.x; i < total; i += stride) {
    const int row = (int)(i >> 12);              // DIMV = 4096
    out[i] = xhat[i] * norms[row] + g4[i] * (QJL_SCALE * resnorm[row]);
  }
}

__global__ void conv_f32_f16(const float* __restrict__ in, _Float16* __restrict__ out, size_t n) {
  const size_t stride = (size_t)gridDim.x * blockDim.x;
  for (size_t i = (size_t)blockIdx.x * blockDim.x + threadIdx.x; i < n; i += stride)
    out[i] = (_Float16)in[i];
}

// LDS-tiled transpose: out[c][r] = (f16) in[r][c], DIMV x DIMV.
// 32x32 tiles, 256 threads (8 rows of 32), LDS padded to kill bank conflicts.
__global__ void __launch_bounds__(256)
transpose_f32_f16(const float* __restrict__ in, _Float16* __restrict__ out) {
  __shared__ _Float16 tile[32][33];
  const int tx = threadIdx.x & 31;
  const int ty = threadIdx.x >> 5;              // 0..7
  const int rBase = blockIdx.y * 32;
  const int cBase = blockIdx.x * 32;
  for (int i = 0; i < 32; i += 8) {
    const int r = rBase + ty + i;
    tile[ty + i][tx] = (_Float16)in[(size_t)r * DIMV + cBase + tx];
  }
  __syncthreads();
  for (int i = 0; i < 32; i += 8) {
    const int c = cBase + ty + i;
    out[(size_t)c * DIMV + rBase + tx] = tile[tx][ty + i];
  }
}

// ---------------------------------------------------------------------------
extern "C" void kernel_launch(void* const* d_in, const int* in_sizes, int n_in,
                              void* d_out, int out_size, void* d_ws, size_t ws_size,
                              hipStream_t stream) {
  const float* x    = (const float*)d_in[0];   // (8192, 4096)
  const float* Pi   = (const float*)d_in[1];   // (4096, 4096)
  const float* cent = (const float*)d_in[2];   // (4,)
  const float* S    = (const float*)d_in[3];   // (4096, 4096)
  float* out = (float*)d_out;

  // ---- carve workspace (256B-aligned partitions) ----
  char* w = (char*)d_ws;
  auto carve = [&](size_t bytes) -> void* {
    void* p = (void*)w;
    w += (bytes + 255) & ~(size_t)255;
    return p;
  };
  float*    norms   = (float*)carve((size_t)NROWSV * 4);
  float*    resnorm = (float*)carve((size_t)NROWSV * 4);
  float*    var     = (float*)carve((size_t)DIMV * 4);
  int*      mask    = (int*)  carve((size_t)DIMV * 4);
  _Float16* Pi16    = (_Float16*)carve((size_t)DIMV * DIMV * 2);
  _Float16* PiT16   = (_Float16*)carve((size_t)DIMV * DIMV * 2);
  _Float16* S16     = (_Float16*)carve((size_t)DIMV * DIMV * 2);
  _Float16* ST16    = (_Float16*)carve((size_t)DIMV * DIMV * 2);
  _Float16* bufH    = (_Float16*)carve((size_t)NROWSV * DIMV * 2); // xu -> yhat -> res -> signs
  float*    F1      = (float*)carve((size_t)NROWSV * DIMV * 4);    // y -> xhat (kept to end)
  float*    F2      = (float*)carve((size_t)NROWSV * DIMV * 4);    // proj -> signs@S

  const dim3 gemmGrid(DIMV / 64, NROWSV / 256);
  const dim3 gemmBlk(256);
  const dim3 trGrid(DIMV / 32, DIMV / 32);

  // fp16 operand prep
  conv_f32_f16     <<<4096, 256, 0, stream>>>(Pi, Pi16, (size_t)DIMV * DIMV);
  transpose_f32_f16<<<trGrid, 256, 0, stream>>>(Pi, PiT16);
  conv_f32_f16     <<<4096, 256, 0, stream>>>(S,  S16,  (size_t)DIMV * DIMV);
  transpose_f32_f16<<<trGrid, 256, 0, stream>>>(S,  ST16);

  // norms + x_unit (fp16)
  rownorm_kernel<<<NROWSV, 256, 0, stream>>>(x, norms, bufH);

  // outlier channels
  colvar_kernel      <<<DIMV / 256, 256, 0, stream>>>(x, var);
  outlier_mask_kernel<<<DIMV / 256, 256, 0, stream>>>(var, mask);

  // GEMM1: y = x_unit @ Pi^T          (Bt = Pi)
  gemm_wmma_f16<<<gemmGrid, gemmBlk, 0, stream>>>(bufH, Pi16, F1, NROWSV, DIMV, DIMV);

  // quantize y -> yhat (fp16, overwrites x_unit buffer)
  quantize_kernel<<<8192, 256, 0, stream>>>(F1, cent, bufH);

  // GEMM2: xhat = yhat @ Pi           (Bt = Pi^T)
  gemm_wmma_f16<<<gemmGrid, gemmBlk, 0, stream>>>(bufH, PiT16, F1, NROWSV, DIMV, DIMV);

  // residual (fp16, overwrites yhat buffer) + residual norms
  residual_kernel<<<NROWSV, 256, 0, stream>>>(x, F1, norms, mask, bufH, resnorm);

  // GEMM3: proj = residual @ S^T      (Bt = S)
  gemm_wmma_f16<<<gemmGrid, gemmBlk, 0, stream>>>(bufH, S16, F2, NROWSV, DIMV, DIMV);

  // signs (fp16, overwrites residual buffer)
  sign_kernel<<<8192, 256, 0, stream>>>(F2, bufH);

  // GEMM4: q = signs @ S              (Bt = S^T)
  gemm_wmma_f16<<<gemmGrid, gemmBlk, 0, stream>>>(bufH, ST16, F2, NROWSV, DIMV, DIMV);

  // out = xhat*norm + q*(QJL_SCALE*resnorm)
  combine_kernel<<<8192, 256, 0, stream>>>(F1, F2, norms, resnorm, out);
}
